// LinearModel_63247688401163
// MI455X (gfx1250) — compile-verified, matrix operand order checked by reading, not run on hardware
//
#include <hip/hip_runtime.h>

typedef __attribute__((ext_vector_type(16))) __bf16 v16bf;
typedef __attribute__((ext_vector_type(8)))  float  v8f;

#define BB 32
#define TT 128
#define DD 128
#define VV 32000
#define KTOT (TT * DD)   // 16384
#define LXLD 520         // LDS row stride in bf16 (512 + 8 pad -> bank-conflict-free b128)
#define NSLC 8           // split-K slices of 2048

static __device__ __forceinline__ int imin(int a, int b) { return a < b ? a : b; }

// Assemble a 16-element bf16 fragment from two 16-byte groups (elements 0..7, 8..15).
static __device__ __forceinline__ v16bf ld16bf(const __bf16* p0, const __bf16* p1) {
  uint4 lo = *(const uint4*)p0;
  uint4 hi = *(const uint4*)p1;
  v16bf r;
  __builtin_memcpy(&r, &lo, 16);
  __builtin_memcpy((char*)&r + 16, &hi, 16);
  return r;
}

// Load 8+8 f32 and convert to a bf16 fragment (operands live in HBM as f32; cvt is free vs BW).
static __device__ __forceinline__ v16bf ld16bf_f32(const float* p0, const float* p1) {
  float4 a0 = *(const float4*)p0;
  float4 a1 = *(const float4*)(p0 + 4);
  float4 b0 = *(const float4*)p1;
  float4 b1 = *(const float4*)(p1 + 4);
  v16bf r;
  r[0]  = (__bf16)a0.x; r[1]  = (__bf16)a0.y; r[2]  = (__bf16)a0.z; r[3]  = (__bf16)a0.w;
  r[4]  = (__bf16)a1.x; r[5]  = (__bf16)a1.y; r[6]  = (__bf16)a1.z; r[7]  = (__bf16)a1.w;
  r[8]  = (__bf16)b0.x; r[9]  = (__bf16)b0.y; r[10] = (__bf16)b0.z; r[11] = (__bf16)b0.w;
  r[12] = (__bf16)b1.x; r[13] = (__bf16)b1.y; r[14] = (__bf16)b1.z; r[15] = (__bf16)b1.w;
  return r;
}

static __device__ __forceinline__ v8f wmma_bf16(v16bf a, v16bf b, v8f c) {
  return __builtin_amdgcn_wmma_f32_16x16x32_bf16(false, a, false, b, (short)0, c, false, false);
}

// ---------------- workspace zero (Y slices must start at 0 every call) ----------------
__global__ void k_zero(float4* __restrict__ p, int n4) {
  int i  = blockIdx.x * blockDim.x + threadIdx.x;
  int st = gridDim.x * blockDim.x;
  float4 z; z.x = 0.f; z.y = 0.f; z.z = 0.f; z.w = 0.f;
  for (; i < n4; i += st) p[i] = z;
}

// ---------------- embedding gather + f32->bf16 ----------------
__global__ void k_gather(const int* __restrict__ tok, const float* __restrict__ emb,
                         __bf16* __restrict__ xbf) {
  const int n4 = BB * TT * DD / 4;
  int i  = blockIdx.x * blockDim.x + threadIdx.x;
  int st = gridDim.x * blockDim.x;
  for (; i < n4; i += st) {
    int e   = i << 2;
    int b   = e >> 14;           // / (T*D)
    int rem = e & (KTOT - 1);
    int t   = rem >> 7;          // / D
    int d   = rem & (DD - 1);
    int tk  = tok[b * TT + t];
    float4 v = *(const float4*)(emb + (size_t)tk * DD + d);
    union { __bf16 h[4]; uint2 u; } pk;
    pk.h[0] = (__bf16)v.x; pk.h[1] = (__bf16)v.y;
    pk.h[2] = (__bf16)v.z; pk.h[3] = (__bf16)v.w;
    *(uint2*)(xbf + e) = pk.u;
  }
}

// ---------------- masked big GEMM: Y = X @ W1^T (block-causal), split-K ----------------
// grid.x = t_out (128), grid.y = K slice (8). 8 waves/block, each wave owns a 16-col N-tile.
__global__ __launch_bounds__(256)
void k_gemm1(const __bf16* __restrict__ xbf, const float* __restrict__ w1,
             float* __restrict__ yslc) {
  const int t    = blockIdx.x;
  const int s    = blockIdx.y;
  const int kmax = (t + 1) * DD;              // causal mask: only K < (t+1)*128 contributes
  const int kb   = s * 2048;
  const int ke   = imin(kb + 2048, kmax);
  if (kb >= ke) return;

  __shared__ __attribute__((aligned(16))) __bf16 lx[32 * LXLD];

  const int tid  = threadIdx.x;
  const int wave = tid >> 5;
  const int lane = tid & 31;
  const int n0   = t * DD + wave * 16;
  const int mrow = lane & 15;
  const int ksel = (lane >> 4) << 3;          // 0 or 8: ISA A/B bf16 lane K-grouping

  v8f c0 = {};  // M rows 0..15
  v8f c1 = {};  // M rows 16..31

  for (int kc = kb; kc < ke; kc += 512) {
    const int kchunk = imin(512, ke - kc);    // multiple of 128
    {   // cooperative stage of X[:, kc..kc+kchunk) into LDS
      const int row = tid >> 3;               // 0..31
      const int c8  = tid & 7;
      const int nv  = kchunk >> 3;            // 16B vectors per row
      const uint4* g = (const uint4*)(xbf + (size_t)row * KTOT + kc);
      uint4* l       = (uint4*)(lx + row * LXLD);
      for (int c = c8; c < nv; c += 8) l[c] = g[c];
    }
    __syncthreads();
    for (int kk = 0; kk < kchunk; kk += 32) {
      const __bf16* ap = lx + mrow * LXLD + kk + ksel;
      v16bf a0 = ld16bf(ap, ap + 16);                       // ds_load_b128 x2
      v16bf a1 = ld16bf(ap + 16 * LXLD, ap + 16 * LXLD + 16);
      const float* bp = w1 + (size_t)(n0 + mrow) * KTOT + kc + kk + ksel;
      v16bf bf = ld16bf_f32(bp, bp + 16);                   // stream W1 f32, cvt->bf16
      c0 = wmma_bf16(a0, bf, c0);
      c1 = wmma_bf16(a1, bf, c1);
    }
    __syncthreads();
  }

  // Deterministic split-K: each slice writes its own buffer (no fp atomics).
  float* yp   = yslc + (size_t)s * (BB * KTOT);
  const int n  = lane & 15;
  const int mb = (lane >> 4) << 3;
#pragma unroll
  for (int r = 0; r < 8; ++r) {
    yp[(size_t)(mb + r) * KTOT + n0 + n]      = c0[r];
    yp[(size_t)(16 + mb + r) * KTOT + n0 + n] = c1[r];
  }
}

// ---------------- reduce split-K slices + ReLU + cvt->bf16 ----------------
__global__ void k_combine(const float4* __restrict__ yslc, __bf16* __restrict__ ybf) {
  const int n4 = BB * KTOT / 4;
  int i  = blockIdx.x * blockDim.x + threadIdx.x;
  int st = gridDim.x * blockDim.x;
  for (; i < n4; i += st) {
    float4 acc = yslc[i];
#pragma unroll
    for (int s = 1; s < NSLC; ++s) {
      float4 v = yslc[(size_t)s * n4 + i];
      acc.x += v.x; acc.y += v.y; acc.z += v.z; acc.w += v.w;
    }
    acc.x = fmaxf(acc.x, 0.f); acc.y = fmaxf(acc.y, 0.f);
    acc.z = fmaxf(acc.z, 0.f); acc.w = fmaxf(acc.w, 0.f);
    union { __bf16 h[4]; uint2 u; } pk;
    pk.h[0] = (__bf16)acc.x; pk.h[1] = (__bf16)acc.y;
    pk.h[2] = (__bf16)acc.z; pk.h[3] = (__bf16)acc.w;
    *(uint2*)(ybf + (size_t)i * 4) = pk.u;
  }
}

// ---------------- X2 = relu(Y @ W2^T + b2), [4096x128] ----------------
__global__ __launch_bounds__(256)
void k_gemm2(const __bf16* __restrict__ ybf, const float* __restrict__ w2,
             const float* __restrict__ b2, __bf16* __restrict__ x2) {
  const int wave = threadIdx.x >> 5;
  const int lane = threadIdx.x & 31;
  const int tile = blockIdx.x * 8 + wave;     // 2048 tiles
  const int m0   = (tile >> 3) * 16;
  const int n0   = (tile & 7) * 16;
  const int mrow = lane & 15;
  const int ksel = (lane >> 4) << 3;
  v8f c = {};
#pragma unroll
  for (int kk = 0; kk < DD; kk += 32) {
    const __bf16* ap = ybf + (size_t)(m0 + mrow) * DD + kk + ksel;
    v16bf a = ld16bf(ap, ap + 16);
    const float* bp = w2 + (size_t)(n0 + mrow) * DD + kk + ksel;
    v16bf b = ld16bf_f32(bp, bp + 16);
    c = wmma_bf16(a, b, c);
  }
  const int n    = lane & 15;
  const int mb   = (lane >> 4) << 3;
  const float bv = b2[n0 + n];
#pragma unroll
  for (int r = 0; r < 8; ++r) {
    float v = fmaxf(c[r] + bv, 0.f);
    x2[(size_t)(m0 + mb + r) * DD + n0 + n] = (__bf16)v;
  }
}

// ---------------- out = X2 @ Wout^T + bout, [4096x32000] f32 ----------------
// grid (2000 N-tiles, 32); 8 waves/block share one N-tile => Wout tile reuse in L0/L2.
__global__ __launch_bounds__(256)
void k_gemm3(const __bf16* __restrict__ x2, const float* __restrict__ wout,
             const float* __restrict__ bout, float* __restrict__ out) {
  const int wave = threadIdx.x >> 5;
  const int lane = threadIdx.x & 31;
  const int n0   = blockIdx.x * 16;
  const int m0   = (blockIdx.y * 8 + wave) * 16;
  const int mrow = lane & 15;
  const int ksel = (lane >> 4) << 3;
  v8f c = {};
#pragma unroll
  for (int kk = 0; kk < DD; kk += 32) {
    const __bf16* ap = x2 + (size_t)(m0 + mrow) * DD + kk + ksel;
    v16bf a = ld16bf(ap, ap + 16);
    const float* bp = wout + (size_t)(n0 + mrow) * DD + kk + ksel;
    v16bf b = ld16bf_f32(bp, bp + 16);
    c = wmma_bf16(a, b, c);
  }
  const int n    = lane & 15;
  const int mb   = (lane >> 4) << 3;
  const float bv = bout[n0 + n];
#pragma unroll
  for (int r = 0; r < 8; ++r) {
    out[(size_t)(m0 + mb + r) * VV + n0 + n] = c[r] + bv;
  }
}

extern "C" void kernel_launch(void* const* d_in, const int* in_sizes, int n_in,
                              void* d_out, int out_size, void* d_ws, size_t ws_size,
                              hipStream_t stream) {
  const int*   tokens = (const int*)d_in[0];
  const float* emb    = (const float*)d_in[1];
  const float* W1     = (const float*)d_in[2];
  const float* W2     = (const float*)d_in[3];
  const float* b2     = (const float*)d_in[4];
  const float* Wout   = (const float*)d_in[5];
  const float* bout   = (const float*)d_in[6];
  float* out = (float*)d_out;

  char* ws = (char*)d_ws;
  __bf16* Xbf  = (__bf16*)(ws);                          //  1 MB: [32,16384] bf16
  float*  Yslc = (float*)(ws + (1u << 20));              // 16 MB: 8 x [32,16384] f32
  __bf16* Ybf  = (__bf16*)(ws + (17u << 20));            //  1 MB: [4096,128] bf16
  __bf16* X2   = (__bf16*)(ws + (18u << 20));            //  1 MB: [4096,128] bf16

  // zero the split-K slice buffers (deterministic accumulation, no fp atomics)
  k_zero<<<4096, 256, 0, stream>>>((float4*)Yslc, NSLC * BB * KTOT / 4);
  k_gather<<<512, 256, 0, stream>>>(tokens, emb, Xbf);
  k_gemm1<<<dim3(TT, NSLC), 256, 0, stream>>>(Xbf, W1, Yslc);
  k_combine<<<512, 256, 0, stream>>>((const float4*)Yslc, Ybf);
  k_gemm2<<<256, 256, 0, stream>>>(Ybf, W2, b2, X2);
  k_gemm3<<<dim3(VV / 16, 32), 256, 0, stream>>>(X2, Wout, bout, out);
}
